// LibrarianCurator_84885733638189
// MI455X (gfx1250) — matrix operand order repaired and checked by reading, not compiled, not run on hardware
//
#include <hip/hip_runtime.h>
#include <hip/hip_bf16.h>

// ---------------------------------------------------------------------------
// MI455X / gfx1250 implementation.
// Pipeline: S@W1+b1 -> LayerNorm -> GELU(exact) -> @W2+b2 -> L2 normalize
//           -> sim = E @ E^T (symmetric, computed upper-triangle only)
//           -> is_duplicate[j] = any_{i<j} sim[i,j] > 0.85
// All GEMMs use v_wmma_f32_16x16x32_f16 (wave32), f32 accumulation.
// ---------------------------------------------------------------------------

typedef __attribute__((ext_vector_type(16))) _Float16 v16h;
typedef __attribute__((ext_vector_type(8)))  _Float16 v8h;
typedef __attribute__((ext_vector_type(4)))  _Float16 v4h;
typedef __attribute__((ext_vector_type(8)))  float    v8f;

static constexpr int   NROW   = 8192;
static constexpr int   DM     = 1024;
static constexpr float THRESH = 0.85f;

// ---------------- elementwise f32 -> f16 (row-major copy) -------------------
__global__ __launch_bounds__(256) void k_f32_to_f16(const float* __restrict__ in,
                                                    _Float16* __restrict__ out,
                                                    int n4) {
  int i = blockIdx.x * blockDim.x + threadIdx.x;
  if (i >= n4) return;
  float4 v = reinterpret_cast<const float4*>(in)[i];
  v4h o;
  o[0] = (_Float16)v.x; o[1] = (_Float16)v.y;
  o[2] = (_Float16)v.z; o[3] = (_Float16)v.w;
  reinterpret_cast<v4h*>(out)[i] = o;
}

// ---------------- transpose 1024x1024 f32 -> f16 (LDS tiled) ----------------
// out[n*DM + k] = in[k*DM + n]; block (32,8), grid (32,32)
__global__ __launch_bounds__(256) void k_transpose_to_f16(const float* __restrict__ in,
                                                          _Float16* __restrict__ out) {
  __shared__ float tile[32][33];
  const int x = blockIdx.x * 32 + threadIdx.x;
  const int ybase = blockIdx.y * 32;
#pragma unroll
  for (int j = 0; j < 32; j += 8) {
    tile[threadIdx.y + j][threadIdx.x] = in[(size_t)(ybase + threadIdx.y + j) * DM + x];
  }
  __syncthreads();
  const int xo = blockIdx.y * 32 + threadIdx.x;
  const int yobase = blockIdx.x * 32;
#pragma unroll
  for (int j = 0; j < 32; j += 8) {
    out[(size_t)(yobase + threadIdx.y + j) * DM + xo] =
        (_Float16)tile[threadIdx.x][threadIdx.y + j];
  }
}

// ---------------- WMMA fragment loaders -------------------------------------
// A 16x32 f16 fragment: lane<16 holds row M=lane, K = {0..7} U {16..23};
// lane>=16 holds M=lane-16, K = {8..15} U {24..31}.  (ISA 7.12.2)
__device__ __forceinline__ v16h load_a_frag(const _Float16* __restrict__ base,
                                            int row, int K, int kb, int hi, int mrow) {
  const _Float16* p = base + (size_t)(row + mrow) * K + kb + hi * 8;
  v8h lo = *reinterpret_cast<const v8h*>(p);
  v8h hh = *reinterpret_cast<const v8h*>(p + 16);
  return __builtin_shufflevector(lo, hh, 0, 1, 2, 3, 4, 5, 6, 7,
                                 8, 9, 10, 11, 12, 13, 14, 15);
}
// B 32x16 f16 fragment from B^T (row-major [N][K]): lane<16 holds col=lane,
// K = {0..7} U {16..23}; lane>=16 holds col=lane-16, K = {8..15} U {24..31}.
__device__ __forceinline__ v16h load_b_frag(const _Float16* __restrict__ baseT,
                                            int col, int K, int kb, int hi, int mrow) {
  const _Float16* p = baseT + (size_t)(col + mrow) * K + kb + hi * 8;
  v8h lo = *reinterpret_cast<const v8h*>(p);
  v8h hh = *reinterpret_cast<const v8h*>(p + 16);
  return __builtin_shufflevector(lo, hh, 0, 1, 2, 3, 4, 5, 6, 7,
                                 8, 9, 10, 11, 12, 13, 14, 15);
}

// ---------------- GEMM: C[M x N] = A[M x K] * B, Bt = B^T [N x K], + bias ---
// block = 256 threads = 8 waves (2 rows x 4 cols of 32x32 wave tiles)
__global__ __launch_bounds__(256) void k_gemm_f16_bias(const _Float16* __restrict__ A,
                                                       const _Float16* __restrict__ Bt,
                                                       const float* __restrict__ bias,
                                                       float* __restrict__ C,
                                                       int Ncols, int K) {
  const int lane = threadIdx.x & 31;
  const int wave = threadIdx.x >> 5;
  const int hi = lane >> 4;
  const int mrow = lane & 15;
  const int rbase = blockIdx.y * 64 + (wave >> 2) * 32;
  const int cbase = blockIdx.x * 128 + (wave & 3) * 32;

  const v8f vzero = {0.f, 0.f, 0.f, 0.f, 0.f, 0.f, 0.f, 0.f};
  v8f acc[2][2];
#pragma unroll
  for (int i = 0; i < 2; ++i)
#pragma unroll
    for (int j = 0; j < 2; ++j) acc[i][j] = vzero;

  for (int kb = 0; kb < K; kb += 32) {
    v16h afr[2], bfr[2];
#pragma unroll
    for (int t = 0; t < 2; ++t) {
      afr[t] = load_a_frag(A, rbase + t * 16, K, kb, hi, mrow);
      bfr[t] = load_b_frag(Bt, cbase + t * 16, K, kb, hi, mrow);
    }
#pragma unroll
    for (int ti = 0; ti < 2; ++ti)
#pragma unroll
      for (int tj = 0; tj < 2; ++tj)
        acc[ti][tj] = __builtin_amdgcn_wmma_f32_16x16x32_f16(
            false, afr[ti], false, bfr[tj], (short)0, acc[ti][tj], false, false);
  }
  // C/D layout: VGPR r, lanes 0-15 -> M=r, N=lane; lanes 16-31 -> M=r+8.
#pragma unroll
  for (int ti = 0; ti < 2; ++ti) {
#pragma unroll
    for (int tj = 0; tj < 2; ++tj) {
      const int gc = cbase + tj * 16 + mrow;
      const float bv = bias[gc];
#pragma unroll
      for (int r = 0; r < 8; ++r) {
        const int gr = rbase + ti * 16 + hi * 8 + r;
        C[(size_t)gr * Ncols + gc] = acc[ti][tj][r] + bv;
      }
    }
  }
}

// ---------------- LayerNorm + exact GELU -> f16 -----------------------------
__global__ __launch_bounds__(256) void k_ln_gelu(const float* __restrict__ H,
                                                 const float* __restrict__ gamma,
                                                 const float* __restrict__ beta,
                                                 _Float16* __restrict__ out) {
  __shared__ float s1[256], s2[256];
  const int row = blockIdx.x, t = threadIdx.x;
  const float4 x = reinterpret_cast<const float4*>(H + (size_t)row * DM)[t];
  s1[t] = x.x + x.y + x.z + x.w;
  s2[t] = x.x * x.x + x.y * x.y + x.z * x.z + x.w * x.w;
  __syncthreads();
  for (int off = 128; off > 0; off >>= 1) {
    if (t < off) { s1[t] += s1[t + off]; s2[t] += s2[t + off]; }
    __syncthreads();
  }
  const float mu = s1[0] * (1.0f / DM);
  const float var = s2[0] * (1.0f / DM) - mu * mu;
  const float rstd = rsqrtf(var + 1e-5f);
  const float4 g = reinterpret_cast<const float4*>(gamma)[t];
  const float4 b = reinterpret_cast<const float4*>(beta)[t];
  const float xv[4] = {x.x, x.y, x.z, x.w};
  const float gv[4] = {g.x, g.y, g.z, g.w};
  const float bv[4] = {b.x, b.y, b.z, b.w};
  v4h o;
#pragma unroll
  for (int i = 0; i < 4; ++i) {
    const float y = (xv[i] - mu) * rstd * gv[i] + bv[i];
    const float ge = 0.5f * y * (1.0f + erff(y * 0.70710678118654752f));
    o[i] = (_Float16)ge;
  }
  reinterpret_cast<v4h*>(out + (size_t)row * DM)[t] = o;
}

// ---------------- row L2 normalize -> f16 -----------------------------------
__global__ __launch_bounds__(256) void k_l2norm(const float* __restrict__ E,
                                                _Float16* __restrict__ out) {
  __shared__ float s2[256];
  const int row = blockIdx.x, t = threadIdx.x;
  const float4 x = reinterpret_cast<const float4*>(E + (size_t)row * DM)[t];
  s2[t] = x.x * x.x + x.y * x.y + x.z * x.z + x.w * x.w;
  __syncthreads();
  for (int off = 128; off > 0; off >>= 1) {
    if (t < off) s2[t] += s2[t + off];
    __syncthreads();
  }
  const float inv = 1.0f / fmaxf(sqrtf(s2[0]), 1e-12f);
  v4h o;
  o[0] = (_Float16)(x.x * inv); o[1] = (_Float16)(x.y * inv);
  o[2] = (_Float16)(x.z * inv); o[3] = (_Float16)(x.w * inv);
  reinterpret_cast<v4h*>(out + (size_t)row * DM)[t] = o;
}

// ---------------- zero-init duplicate flags ---------------------------------
__global__ void k_zero(float* __restrict__ p, int n) {
  int i = blockIdx.x * blockDim.x + threadIdx.x;
  if (i < n) p[i] = 0.0f;
}

// ---------------- sim = E @ E^T (symmetric) + duplicate flags ---------------
// Wave tiles strictly below the diagonal are skipped; upper/diagonal tiles
// are staged in LDS and stored twice (direct + mirrored), both float4-coalesced.
__global__ __launch_bounds__(256) void k_sim(const _Float16* __restrict__ E,
                                             float* __restrict__ sim,
                                             float* __restrict__ dup) {
  __shared__ float mt_all[8][32][33];
  const int rb0 = blockIdx.y * 64;
  const int cb0 = blockIdx.x * 128;
  if (cb0 + 128 <= rb0) return;  // whole block strictly below diagonal

  const int lane = threadIdx.x & 31;
  const int wave = threadIdx.x >> 5;
  const int hi = lane >> 4;
  const int mrow = lane & 15;
  const int rbase = rb0 + (wave >> 2) * 32;
  const int cbase = cb0 + (wave & 3) * 32;
  if (cbase + 32 <= rbase) return;  // wave tile strictly below diagonal

  const v8f vzero = {0.f, 0.f, 0.f, 0.f, 0.f, 0.f, 0.f, 0.f};
  v8f acc[2][2];
#pragma unroll
  for (int i = 0; i < 2; ++i)
#pragma unroll
    for (int j = 0; j < 2; ++j) acc[i][j] = vzero;

  for (int kb = 0; kb < DM; kb += 32) {
    v16h afr[2], bfr[2];
#pragma unroll
    for (int t = 0; t < 2; ++t) {
      afr[t] = load_a_frag(E, rbase + t * 16, DM, kb, hi, mrow);
      bfr[t] = load_b_frag(E, cbase + t * 16, DM, kb, hi, mrow);
    }
#pragma unroll
    for (int ti = 0; ti < 2; ++ti)
#pragma unroll
      for (int tj = 0; tj < 2; ++tj)
        acc[ti][tj] = __builtin_amdgcn_wmma_f32_16x16x32_f16(
            false, afr[ti], false, bfr[tj], (short)0, acc[ti][tj], false, false);
  }

  // Stage the 32x32 tile into this wave's private LDS slice.
  float(*mt)[33] = mt_all[wave];
#pragma unroll
  for (int ti = 0; ti < 2; ++ti)
#pragma unroll
    for (int tj = 0; tj < 2; ++tj)
#pragma unroll
      for (int r = 0; r < 8; ++r)
        mt[ti * 16 + hi * 8 + r][tj * 16 + mrow] = acc[ti][tj][r];

  // Intra-wave cross-lane LDS handoff: fence with the CDNA5 split DS counter.
  // (No __syncthreads: lower-triangle waves may have exited already.)
  asm volatile("s_wait_dscnt 0" ::: "memory");

  const int colGrp = lane & 7;   // 4-column group within the tile
  const int rowSel = lane >> 3;  // 0..3: row within each group of 4 rows

  // Direct store (rows rbase.., cols cbase..) + duplicate flags (i < j).
#pragma unroll
  for (int it = 0; it < 8; ++it) {
    const int lr = it * 4 + rowSel;
    const int gr = rbase + lr;
    const int gc0 = cbase + colGrp * 4;
    float4 v;
    v.x = mt[lr][colGrp * 4 + 0];
    v.y = mt[lr][colGrp * 4 + 1];
    v.z = mt[lr][colGrp * 4 + 2];
    v.w = mt[lr][colGrp * 4 + 3];
    if (gr < gc0 + 0 && v.x > THRESH) dup[gc0 + 0] = 1.0f;
    if (gr < gc0 + 1 && v.y > THRESH) dup[gc0 + 1] = 1.0f;
    if (gr < gc0 + 2 && v.z > THRESH) dup[gc0 + 2] = 1.0f;
    if (gr < gc0 + 3 && v.w > THRESH) dup[gc0 + 3] = 1.0f;
    *reinterpret_cast<float4*>(sim + (size_t)gr * NROW + gc0) = v;
  }
  // Mirrored store (rows cbase.., cols rbase..), transposed via LDS.
#pragma unroll
  for (int it = 0; it < 8; ++it) {
    const int lc = it * 4 + rowSel;
    const int orow = cbase + lc;
    float4 v;
    v.x = mt[colGrp * 4 + 0][lc];
    v.y = mt[colGrp * 4 + 1][lc];
    v.z = mt[colGrp * 4 + 2][lc];
    v.w = mt[colGrp * 4 + 3][lc];
    *reinterpret_cast<float4*>(sim + (size_t)orow * NROW + rbase + colGrp * 4) = v;
  }
}

// ---------------------------------------------------------------------------
extern "C" void kernel_launch(void* const* d_in, const int* in_sizes, int n_in,
                              void* d_out, int out_size, void* d_ws, size_t ws_size,
                              hipStream_t stream) {
  const float* S     = (const float*)d_in[0];
  const float* W1    = (const float*)d_in[1];
  const float* b1    = (const float*)d_in[2];
  const float* gamma = (const float*)d_in[3];
  const float* beta  = (const float*)d_in[4];
  const float* W2    = (const float*)d_in[5];
  const float* b2    = (const float*)d_in[6];

  float* sim = (float*)d_out;
  float* dup = sim + (size_t)NROW * NROW;  // is_duplicate as 0.0/1.0

  // Workspace layout (84 MB total):
  char* ws = (char*)d_ws;
  _Float16* W1t = (_Float16*)(ws);                        //  2 MB  W1^T f16
  _Float16* W2t = (_Float16*)(ws + (size_t)(2)  * 1048576); //  2 MB  W2^T f16
  _Float16* Sh  = (_Float16*)(ws + (size_t)(4)  * 1048576); // 16 MB  summaries f16
  float*    Hf  = (float*)   (ws + (size_t)(20) * 1048576); // 32 MB  h f32
  _Float16* Gh  = (_Float16*)(ws + (size_t)(52) * 1048576); // 16 MB  gelu(ln(h)) f16
  float*    Ef  = (float*)   (ws + (size_t)(20) * 1048576); // 32 MB  enc f32 (reuses Hf)
  _Float16* Eh  = (_Float16*)(ws + (size_t)(68) * 1048576); // 16 MB  normalized enc f16

  // 0) zero duplicate flags (output tail is poisoned by harness)
  k_zero<<<(NROW + 255) / 256, 256, 0, stream>>>(dup, NROW);

  // 1) precision conversions / weight transposes
  k_f32_to_f16<<<(NROW * DM / 4 + 255) / 256, 256, 0, stream>>>(S, Sh, NROW * DM / 4);
  k_transpose_to_f16<<<dim3(32, 32), dim3(32, 8), 0, stream>>>(W1, W1t);
  k_transpose_to_f16<<<dim3(32, 32), dim3(32, 8), 0, stream>>>(W2, W2t);

  // 2) h = S @ W1 + b1
  k_gemm_f16_bias<<<dim3(DM / 128, NROW / 64), 256, 0, stream>>>(Sh, W1t, b1, Hf, DM, DM);

  // 3) gelu(layernorm(h)) -> f16
  k_ln_gelu<<<NROW, 256, 0, stream>>>(Hf, gamma, beta, Gh);

  // 4) enc = g @ W2 + b2
  k_gemm_f16_bias<<<dim3(DM / 128, NROW / 64), 256, 0, stream>>>(Gh, W2t, b2, Ef, DM, DM);

  // 5) L2 row-normalize -> f16 (16 MB: fits in 192 MB L2 for the sim GEMM)
  k_l2norm<<<NROW, 256, 0, stream>>>(Ef, Eh);

  // 6) sim = E @ E^T (upper triangle computed, mirrored) + duplicate flags
  k_sim<<<dim3(NROW / 128, NROW / 64), 256, 0, stream>>>(Eh, sim, dup);
}
